// AttnSeq2Seq_68496138437035
// MI455X (gfx1250) — matrix operand order repaired and checked by reading, not compiled, not run on hardware
//
#include <hip/hip_runtime.h>
#include <stdint.h>

// ---------------------------------------------------------------------------
// AttnSeq2Seq on MI455X (gfx1250, wave32, WMMA).
//
//  * bf16 WMMA (v_wmma_f32_16x16x32_bf16) everywhere; weights pre-swizzled
//    into B-fragment lane layout (one contiguous 32B load per lane/k-tile).
//  * Encoder: 32 persistent WGs x 384 threads (12 waves); each WG owns a
//    32-row batch tile (MT=32 -> 2 M-tiles per B-fragment, 2x arithmetic
//    intensity on the L2 weight stream), carries h in LDS (bf16) and c in
//    registers across all 336 steps.  Wave w owns h-cols [32w,32w+32) with
//    matching i/f/g/o z-tiles -> lane-local cell update, 2 barriers/step.
//  * proj: Wa (288KB swizzled) staged once into LDS per persistent block via
//    gfx1250 async global->LDS (ASYNCcnt), then 42 M-tiles per block with
//    B-fragments from LDS -> L2 weight traffic 147MB instead of 6.2GB.
//  * Decoder: 32 persistent WGs, attention on VALU (tiny FLOPs), dec_in GEMM
//    + LSTM cell on WMMA, 18 steps.
// ---------------------------------------------------------------------------

#define IN_D   8
#define H      384
#define G4     1536          // 4*H
#define L_SEQ  336
#define HOR    18
#define MT     32            // batch rows per workgroup (two WMMA M tiles)
#define NW     12            // waves per block (384 threads)
#define K0P    416           // layer0 K: 8(x)+384(h)+24 pad -> 13 k-tiles
#define K1P    768           // layer1/dec K: 384+384       -> 24 k-tiles
#define KDINP  416           // dec_in K: 389 pad to 416    -> 13 k-tiles

typedef __attribute__((ext_vector_type(16))) __bf16 v16bf;
typedef __attribute__((ext_vector_type(8)))  float  v8f;
typedef __attribute__((ext_vector_type(4)))  int    v4i;

#if __has_builtin(__builtin_amdgcn_global_load_async_to_lds_b128) && \
    __has_builtin(__builtin_amdgcn_s_wait_asynccnt)
#define USE_ASYNC_LDS 1
typedef __attribute__((address_space(1))) v4i* gv4i_p;   // global int4*
typedef __attribute__((address_space(3))) v4i* lv4i_p;   // LDS int4*
#else
#define USE_ASYNC_LDS 0
#endif

union FragU { v16bf v; uint4 q[2]; };

__device__ __forceinline__ unsigned short f2bf(float f) {
  union { float f; uint32_t u; } v; v.f = f;
  uint32_t u = v.u;
  return (unsigned short)((u + 0x7FFFu + ((u >> 16) & 1u)) >> 16);  // RNE
}
__device__ __forceinline__ float bf2f(unsigned short h) {
  union { uint32_t u; float f; } v; v.u = ((uint32_t)h) << 16;
  return v.f;
}
__device__ __forceinline__ float sigf(float x) { return 1.0f / (1.0f + __expf(-x)); }
__device__ __forceinline__ float tanh_f(float x) {
  x = fminf(fmaxf(x, -15.0f), 15.0f);
  float e = __expf(2.0f * x);
  return (e - 1.0f) / (e + 1.0f);
}
__device__ __forceinline__ v8f zero8() {
  v8f z = {0.f, 0.f, 0.f, 0.f, 0.f, 0.f, 0.f, 0.f};
  return z;
}

// A-fragment (16x32 bf16): lane l -> row m=l&15, K-half kh=8*(l>>4);
// element e<8 -> k0+kh+e, e>=8 -> k0+16+kh+(e-8): two contiguous 16B loads.
__device__ __forceinline__ v16bf load_frag_row(const unsigned short* base,
                                               int stride, int k0, int lane) {
  int m  = lane & 15;
  int kh = (lane >> 4) << 3;
  const unsigned short* p = base + (size_t)m * stride + k0 + kh;
  FragU u;
  u.q[0] = *(const uint4*)(p);
  u.q[1] = *(const uint4*)(p + 16);
  return u.v;
}

// B-fragment (32x16 bf16): pre-swizzled, lane reads 32 contiguous bytes.
__device__ __forceinline__ v16bf load_frag_w(const unsigned short* swz,
                                             int kt, int nt, int Ntiles, int lane) {
  const unsigned short* p = swz + (((size_t)kt * Ntiles + nt) << 9) + (lane << 4);
  FragU u;
  u.q[0] = *(const uint4*)(p);
  u.q[1] = *(const uint4*)(p + 8);
  return u.v;
}

// LSTM cell step for a 32-row batch tile, 12 waves.  Wave w owns h-columns
// [w*32, w*32+32): per k-tile it loads 8 B-fragments (4 gates x 2 subtiles)
// and feeds both M-halves -> 16 WMMAs/k-tile/wave.  c in registers.
template <int KT>
__device__ __forceinline__ void lstm_cell_step(
    const unsigned short* catBase, int catStride,   // A: [x|h] bf16 (LDS)
    const unsigned short* Wswz,                     // swizzled [KT*32 x 1536]
    const float* __restrict__ bih, const float* __restrict__ bhh,
    float (&cReg)[2][2][8],                         // [j][mhalf][r]
    unsigned short* hDstA, int strideA,
    unsigned short* hDstB, int strideB,
    unsigned short* hGlobal, size_t gStride,
    float* hFloat,
    int wave, int lane) {
  v8f acc[4][2][2];                                 // [gate][j][mhalf]
#pragma unroll
  for (int g = 0; g < 4; ++g)
#pragma unroll
    for (int j = 0; j < 2; ++j) { acc[g][j][0] = zero8(); acc[g][j][1] = zero8(); }

#pragma unroll 1
  for (int kt = 0; kt < KT; ++kt) {
    v16bf a0 = load_frag_row(catBase, catStride, kt * 32, lane);
    v16bf a1 = load_frag_row(catBase + 16 * catStride, catStride, kt * 32, lane);
    __builtin_prefetch(Wswz + (((size_t)(kt + 1) * 96) << 9) + (lane << 4), 0, 0);
#pragma unroll
    for (int g = 0; g < 4; ++g) {
#pragma unroll
      for (int j = 0; j < 2; ++j) {
        v16bf b = load_frag_w(Wswz, kt, g * 24 + wave * 2 + j, 96, lane);
        acc[g][j][0] = __builtin_amdgcn_wmma_f32_16x16x32_bf16(
            false, a0, false, b, (short)0, acc[g][j][0], false, false);
        acc[g][j][1] = __builtin_amdgcn_wmma_f32_16x16x32_bf16(
            false, a1, false, b, (short)0, acc[g][j][1], false, false);
      }
    }
  }
  __syncthreads();  // all reads of previous h complete

  int n15 = lane & 15;
  int mh  = (lane >> 4) << 3;
#pragma unroll
  for (int j = 0; j < 2; ++j) {
    int   ncol = wave * 32 + j * 16 + n15;
    float bi = bih[ncol]         + bhh[ncol];
    float bf = bih[H + ncol]     + bhh[H + ncol];
    float bg = bih[2 * H + ncol] + bhh[2 * H + ncol];
    float bo = bih[3 * H + ncol] + bhh[3 * H + ncol];
#pragma unroll
    for (int mhalf = 0; mhalf < 2; ++mhalf) {
#pragma unroll
      for (int r = 0; r < 8; ++r) {
        int   m  = mhalf * 16 + mh + r;
        float iv = sigf(acc[0][j][mhalf][r] + bi);
        float fv = sigf(acc[1][j][mhalf][r] + bf);
        float gv = tanh_f(acc[2][j][mhalf][r] + bg);
        float ov = sigf(acc[3][j][mhalf][r] + bo);
        float cc = fv * cReg[j][mhalf][r] + iv * gv;
        cReg[j][mhalf][r] = cc;
        float hh = ov * tanh_f(cc);
        unsigned short hb = f2bf(hh);
        hDstA[(size_t)m * strideA + ncol] = hb;
        if (hDstB)   hDstB[(size_t)m * strideB + ncol] = hb;
        if (hGlobal) hGlobal[(size_t)m * gStride + ncol] = hb;
        if (hFloat)  hFloat[m * H + ncol] = hh;
      }
    }
  }
  __syncthreads();  // h visible before next step's reads
}

// ---------------------------------------------------------------------------
// Prep kernels
// ---------------------------------------------------------------------------
__global__ void cvt_f32_bf16(const float* __restrict__ src,
                             unsigned short* __restrict__ dst, int n) {
  int i = blockIdx.x * blockDim.x + threadIdx.x;
  if (i < n) dst[i] = f2bf(src[i]);
}

// Swizzled bf16 B-operand from (optionally concatenated) PyTorch-layout f32
// weights A:[N][Ka], B:[N][Kb]; k >= Ka+Kb is zero padding.
__global__ void swizzle_kernel(const float* __restrict__ A,
                               const float* __restrict__ B,
                               int Ka, int Kb, int N,
                               unsigned short* __restrict__ dst, int total) {
  int id = blockIdx.x * blockDim.x + threadIdx.x;
  if (id >= total) return;
  int e = id & 15, lane = (id >> 4) & 31, tile = id >> 9;
  int Nt = N >> 4;
  int nt = tile % Nt, kt = tile / Nt;
  int kh   = (lane >> 4) << 3;
  int koff = e + (e >= 8 ? 8 : 0) + kh;
  int k = kt * 32 + koff;
  int n = nt * 16 + (lane & 15);
  float v = 0.f;
  if (k < Ka)                v = A[(size_t)n * Ka + k];
  else if (B && k < Ka + Kb) v = B[(size_t)n * Kb + (k - Ka)];
  dst[id] = f2bf(v);
}

// ---------------------------------------------------------------------------
// Encoder: persistent per 32-row batch tile, 2 stacked LSTM layers, 336 steps.
// ---------------------------------------------------------------------------
__global__ __launch_bounds__(384) void encoder_kernel(
    const unsigned short* __restrict__ xbf,     // [B][L][8] bf16
    const unsigned short* __restrict__ W0swz,   // [416 x 1536] swz
    const unsigned short* __restrict__ W1swz,   // [768 x 1536] swz
    const float* __restrict__ bih0, const float* __restrict__ bhh0,
    const float* __restrict__ bih1, const float* __restrict__ bhh1,
    unsigned short* __restrict__ enc_out) {     // [B][L][H] bf16
  __shared__ unsigned short cat0[MT][K0P];      // [x_t | h0 | pad]
  __shared__ unsigned short cat1[MT][K1P];      // [h0  | h1]
  int tid = threadIdx.x, wave = tid >> 5, lane = tid & 31;
  int b0  = blockIdx.x * MT;

  float c0r[2][2][8], c1r[2][2][8];
#pragma unroll
  for (int j = 0; j < 2; ++j)
#pragma unroll
    for (int m = 0; m < 2; ++m)
#pragma unroll
      for (int r = 0; r < 8; ++r) { c0r[j][m][r] = 0.f; c1r[j][m][r] = 0.f; }

  for (int i = tid; i < MT * K0P; i += 384) (&cat0[0][0])[i] = 0;
  for (int i = tid; i < MT * K1P; i += 384) (&cat1[0][0])[i] = 0;
  __syncthreads();

  for (int t = 0; t < L_SEQ; ++t) {
    if (tid < MT * IN_D) {
      int m = tid >> 3, k = tid & 7;
      cat0[m][k] = xbf[((size_t)(b0 + m) * L_SEQ + t) * IN_D + k];
    }
    __syncthreads();
    lstm_cell_step<13>(&cat0[0][0], K0P, W0swz, bih0, bhh0, c0r,
                       &cat0[0][8], K0P, &cat1[0][0], K1P,
                       nullptr, 0, nullptr, wave, lane);
    lstm_cell_step<24>(&cat1[0][0], K1P, W1swz, bih1, bhh1, c1r,
                       &cat1[0][384], K1P, nullptr, 0,
                       enc_out + ((size_t)b0 * L_SEQ + t) * H, (size_t)L_SEQ * H,
                       nullptr, wave, lane);
  }
}

// ---------------------------------------------------------------------------
// proj = enc_out @ Wa^T.  Persistent blocks; Wa swizzled (288KB) staged once
// into LDS via async global->LDS (ASYNCcnt), then each block sweeps its share
// of the 21504 M-tiles with B-fragments served from LDS.
// ---------------------------------------------------------------------------
#define PROJ_GRID 512
__global__ __launch_bounds__(256) void proj_kernel(
    const unsigned short* __restrict__ enc,
    const unsigned short* __restrict__ WaSwz,
    unsigned short* __restrict__ proj, int mtiles) {
  __shared__ unsigned short waLds[H * H];       // 294912 B, fits 320KB LDS
  int tid = threadIdx.x, wave = tid >> 5, lane = tid & 31;

#if USE_ASYNC_LDS
  {
    // builtin signature (from compiler diagnostic): (v4i global*, v4i lds*,
    // i32 offset, i32 cpol).  LDS offset = low 32 bits of generic pointer
    // (see ISA section 10.2).
    gv4i_p g = (gv4i_p)(uintptr_t)(const void*)WaSwz;
    lv4i_p l = (lv4i_p)(uint32_t)(uintptr_t)(void*)&waLds[0];
    for (int i = tid; i < (int)(H * H * 2 / 16); i += 256)
      __builtin_amdgcn_global_load_async_to_lds_b128(g + i, l + i, 0, 0);
    __builtin_amdgcn_s_wait_asynccnt(0);
  }
#else
  {
    const uint4* gs = (const uint4*)WaSwz;
    uint4*       ls = (uint4*)&waLds[0];
    for (int i = tid; i < (int)(H * H * 2 / 16); i += 256) ls[i] = gs[i];
  }
#endif
  __syncthreads();

  for (int mt = blockIdx.x; mt < mtiles; mt += PROJ_GRID) {
    size_t r0 = (size_t)mt * 16;
    const unsigned short* Abase = enc + r0 * H;
    v8f acc[3] = {zero8(), zero8(), zero8()};
#pragma unroll 1
    for (int kt = 0; kt < 12; ++kt) {
      v16bf a = load_frag_row(Abase, H, kt * 32, lane);
      __builtin_prefetch(Abase + (size_t)16 * H + kt * 32, 0, 0);
#pragma unroll
      for (int j = 0; j < 3; ++j) {
        v16bf b = load_frag_w(&waLds[0], kt, wave * 3 + j, 24, lane);
        acc[j] = __builtin_amdgcn_wmma_f32_16x16x32_bf16(
            false, a, false, b, (short)0, acc[j], false, false);
      }
    }
    int n15 = lane & 15, mh = (lane >> 4) << 3;
#pragma unroll
    for (int j = 0; j < 3; ++j) {
      int ncol = (wave * 3 + j) * 16 + n15;
#pragma unroll
      for (int r = 0; r < 8; ++r)
        proj[(r0 + mh + r) * H + ncol] = f2bf(acc[j][r]);
    }
  }
}

// ---------------------------------------------------------------------------
// Decoder: persistent per 32-row batch tile, 18 steps of attention + GEMMs.
// ---------------------------------------------------------------------------
__global__ __launch_bounds__(384) void decoder_kernel(
    const unsigned short* __restrict__ enc,     // [B][L][H] bf16
    const unsigned short* __restrict__ proj,    // [B][L][H] bf16
    const float* __restrict__ fut,              // [B][18][4]
    const unsigned short* __restrict__ WdinSwz, // [416 x 384] swz
    const float* __restrict__ biasDin,          // [384]
    const unsigned short* __restrict__ WdecSwz, // [768 x 1536] swz
    const float* __restrict__ dbih, const float* __restrict__ dbhh,
    const float* __restrict__ outW, const float* __restrict__ outB,
    float* __restrict__ out) {                  // [B][18]
  __shared__ float          hF[MT][H];          // f32 h for attention dots
  __shared__ unsigned short catA[MT][KDINP];    // [y|feat|ctx|pad] bf16
  __shared__ unsigned short catD[MT][K1P];      // [din | h] bf16
  __shared__ float          scores[MT][L_SEQ];
  __shared__ float          yprev[MT];
  int tid = threadIdx.x, wave = tid >> 5, lane = tid & 31;
  int b0  = blockIdx.x * MT;

  float cDr[2][2][8];
#pragma unroll
  for (int j = 0; j < 2; ++j)
#pragma unroll
    for (int m = 0; m < 2; ++m)
#pragma unroll
      for (int r = 0; r < 8; ++r) cDr[j][m][r] = 0.f;

  for (int i = tid; i < MT * KDINP; i += 384) (&catA[0][0])[i] = 0;
  for (int i = tid; i < MT * H; i += 384) {
    int m = i / H, d = i % H;
    unsigned short hb = enc[((size_t)(b0 + m) * L_SEQ + (L_SEQ - 1)) * H + d];
    hF[m][d]         = bf2f(hb);   // h init = encoder final h
    catD[m][384 + d] = hb;
    catD[m][d]       = 0;
  }
  if (tid < MT) yprev[tid] = 0.f;
  __syncthreads();

  for (int t = 0; t < HOR; ++t) {
    // ---- attention scores: proj[b,l,:] . h[b,:] ----
    for (int i = tid; i < MT * L_SEQ; i += 384) {
      int m = i / L_SEQ, l = i % L_SEQ;
      const unsigned short* pr = proj + ((size_t)(b0 + m) * L_SEQ + l) * H;
      float s = 0.f;
      for (int d = 0; d < H; ++d) s += bf2f(pr[d]) * hF[m][d];
      scores[m][l] = s;
    }
    __syncthreads();
    // ---- softmax over L per sample ----
    if (tid < MT) {
      int m = tid;
      float mx = -1e30f;
      for (int l = 0; l < L_SEQ; ++l) mx = fmaxf(mx, scores[m][l]);
      float sum = 0.f;
      for (int l = 0; l < L_SEQ; ++l) {
        float e = __expf(scores[m][l] - mx);
        scores[m][l] = e;
        sum += e;
      }
      float inv = 1.0f / sum;
      for (int l = 0; l < L_SEQ; ++l) scores[m][l] *= inv;
    }
    __syncthreads();
    // ---- ctx = alpha @ enc_out ----
    for (int i = tid; i < MT * H; i += 384) {
      int m = i / H, d = i % H;
      const unsigned short* er = enc + (size_t)(b0 + m) * L_SEQ * H + d;
      float s = 0.f;
      for (int l = 0; l < L_SEQ; ++l) s += scores[m][l] * bf2f(er[(size_t)l * H]);
      catA[m][5 + d] = f2bf(s);
    }
    if (tid < MT * 5) {
      int m = tid / 5, j = tid % 5;
      catA[m][j] = f2bf(j == 0 ? yprev[m]
                               : fut[((size_t)(b0 + m) * HOR + t) * 4 + (j - 1)]);
    }
    __syncthreads();
    // ---- din = relu(catA @ dec_in_W^T + b), N=384 (24 ntiles / 12 waves) ----
    {
      v8f acc[2][2] = {{zero8(), zero8()}, {zero8(), zero8()}};
#pragma unroll 1
      for (int kt = 0; kt < 13; ++kt) {
        v16bf a0 = load_frag_row(&catA[0][0], KDINP, kt * 32, lane);
        v16bf a1 = load_frag_row(&catA[16][0], KDINP, kt * 32, lane);
#pragma unroll
        for (int j = 0; j < 2; ++j) {
          v16bf b = load_frag_w(WdinSwz, kt, wave * 2 + j, 24, lane);
          acc[j][0] = __builtin_amdgcn_wmma_f32_16x16x32_bf16(
              false, a0, false, b, (short)0, acc[j][0], false, false);
          acc[j][1] = __builtin_amdgcn_wmma_f32_16x16x32_bf16(
              false, a1, false, b, (short)0, acc[j][1], false, false);
        }
      }
      __syncthreads();
      int n15 = lane & 15, mh = (lane >> 4) << 3;
#pragma unroll
      for (int j = 0; j < 2; ++j) {
        int   ncol = (wave * 2 + j) * 16 + n15;
        float bb   = biasDin[ncol];
#pragma unroll
        for (int mhalf = 0; mhalf < 2; ++mhalf)
#pragma unroll
          for (int r = 0; r < 8; ++r) {
            float v = fmaxf(acc[j][mhalf][r] + bb, 0.f);
            catD[mhalf * 16 + mh + r][ncol] = f2bf(v);
          }
      }
      __syncthreads();
    }
    // ---- LSTM cell on [din|h], updates hF + catD h region ----
    lstm_cell_step<24>(&catD[0][0], K1P, WdecSwz, dbih, dbhh, cDr,
                       &catD[0][384], K1P, nullptr, 0, nullptr, 0,
                       &hF[0][0], wave, lane);
    // ---- output head ----
    if (tid < MT) {
      int   m = tid;
      float s = outB[0];
      for (int d = 0; d < H; ++d) s += hF[m][d] * outW[d];
      out[(size_t)(b0 + m) * HOR + t] = s;
      yprev[m] = s;
    }
    __syncthreads();
  }
}

// ---------------------------------------------------------------------------
extern "C" void kernel_launch(void* const* d_in, const int* in_sizes, int n_in,
                              void* d_out, int out_size, void* d_ws, size_t ws_size,
                              hipStream_t stream) {
  (void)n_in; (void)out_size; (void)ws_size;
  const float* x     = (const float*)d_in[0];
  const float* fut   = (const float*)d_in[1];
  const float* Wih0  = (const float*)d_in[2];
  const float* Whh0  = (const float*)d_in[3];
  const float* bih0  = (const float*)d_in[4];
  const float* bhh0  = (const float*)d_in[5];
  const float* Wih1  = (const float*)d_in[6];
  const float* Whh1  = (const float*)d_in[7];
  const float* bih1  = (const float*)d_in[8];
  const float* bhh1  = (const float*)d_in[9];
  const float* Wa    = (const float*)d_in[10];
  const float* WdinF = (const float*)d_in[11];
  const float* bdin  = (const float*)d_in[12];
  const float* dWih  = (const float*)d_in[13];
  const float* dWhh  = (const float*)d_in[14];
  const float* dbih  = (const float*)d_in[15];
  const float* dbhh  = (const float*)d_in[16];
  const float* outW  = (const float*)d_in[17];
  const float* outB  = (const float*)d_in[18];
  float* out = (float*)d_out;

  const int Bv = in_sizes[0] / (L_SEQ * IN_D);  // 1024

  // Workspace carve-out (~540 MB): bf16 x, swizzled weights, enc_out, proj.
  char*  ws = (char*)d_ws;
  size_t o  = 0;
  auto alloc = [&](size_t bytes) -> char* {
    char* p = ws + o;
    o += (bytes + 255) & ~(size_t)255;
    return p;
  };
  unsigned short* xbf     = (unsigned short*)alloc((size_t)Bv * L_SEQ * IN_D * 2);
  unsigned short* W0swz   = (unsigned short*)alloc((size_t)K0P * G4 * 2);
  unsigned short* W1swz   = (unsigned short*)alloc((size_t)K1P * G4 * 2);
  unsigned short* WaSwz   = (unsigned short*)alloc((size_t)H * H * 2);
  unsigned short* WdinSwz = (unsigned short*)alloc((size_t)KDINP * H * 2);
  unsigned short* WdecSwz = (unsigned short*)alloc((size_t)K1P * G4 * 2);
  unsigned short* encOut  = (unsigned short*)alloc((size_t)Bv * L_SEQ * H * 2);
  unsigned short* projB   = (unsigned short*)alloc((size_t)Bv * L_SEQ * H * 2);

  {
    int total = Bv * L_SEQ * IN_D;
    cvt_f32_bf16<<<(total + 255) / 256, 256, 0, stream>>>(x, xbf, total);
  }
  {
    int total = K0P * G4;
    swizzle_kernel<<<(total + 255) / 256, 256, 0, stream>>>(
        Wih0, Whh0, IN_D, H, G4, W0swz, total);
  }
  {
    int total = K1P * G4;
    swizzle_kernel<<<(total + 255) / 256, 256, 0, stream>>>(
        Wih1, Whh1, H, H, G4, W1swz, total);
  }
  {
    int total = H * H;
    swizzle_kernel<<<(total + 255) / 256, 256, 0, stream>>>(
        Wa, nullptr, H, 0, H, WaSwz, total);
  }
  {
    int total = KDINP * H;
    swizzle_kernel<<<(total + 255) / 256, 256, 0, stream>>>(
        WdinF, nullptr, 1 + 4 + H, 0, H, WdinSwz, total);
  }
  {
    int total = K1P * G4;
    swizzle_kernel<<<(total + 255) / 256, 256, 0, stream>>>(
        dWih, dWhh, H, H, G4, WdecSwz, total);
  }

  encoder_kernel<<<Bv / MT, 384, 0, stream>>>(
      xbf, W0swz, W1swz, bih0, bhh0, bih1, bhh1, encOut);
  proj_kernel<<<PROJ_GRID, 256, 0, stream>>>(
      encOut, WaSwz, projB, (Bv * L_SEQ) / 16);
  decoder_kernel<<<Bv / MT, 384, 0, stream>>>(
      encOut, projB, fut, WdinSwz, bdin, WdecSwz, dbih, dbhh, outW, outB, out);
}
